// CPSFMemcellFusedReal_3049426780759
// MI455X (gfx1250) — matrix operand order repaired and checked by reading, not compiled, not run on hardware
//
#include <hip/hip_runtime.h>
#include <cmath>

typedef __attribute__((ext_vector_type(16))) _Float16 v16h;
typedef __attribute__((ext_vector_type(8)))  float    v8f;
typedef __attribute__((ext_vector_type(2)))  float    v2f;

#define B_DIM 1024
#define M_DIM 8192
#define N_DIM 32
#define S_DIM 128
#define MCH   32
#define CHUNKS (M_DIM / MCH)             // 256
#define PI_F  3.14159265358979f

// workspace offsets (bytes)
#define WS_PKD   0                                   // M * 16 float4  = 2 MB
#define WS_THP   (M_DIM * 16 * 16)                   // M * S * f16    = 2 MB
#define WS_META  (WS_THP + M_DIM * S_DIM * 2)        // M float4       = 128 KB
#define WS_ZN2   (WS_META + M_DIM * 16)              // B f32          = 4 KB

union AOp { v16h v; _Float16 h[16]; unsigned u[8]; uint4 q4[2]; };

// ---------------------------------------------------------------------------
// Kernel 0: per-m precompute. Writes:
//   pkd[m][kp] = {zj[2kp], zj[2kp+1], b[2kp], b[2kp+1]}  (fused phase-1 B feed)
//   meta[m]    = {w_perp, w_diff, cj = zj.b, |zj|^2}
//   zn2[b]     = |z_b|^2
// ---------------------------------------------------------------------------
__global__ __launch_bounds__(256) void cpsf_precompute_kernel(
    const float* __restrict__ z,
    const float* __restrict__ zj,
    const float* __restrict__ dvec,
    const float* __restrict__ spar,
    const float* __restrict__ sperp,
    float4* __restrict__ pkd,
    float4* __restrict__ meta,
    float*  __restrict__ zn2)
{
    const int m = blockIdx.x * 256 + threadIdx.x;
    if (m < M_DIM) {
        const float4* zr4 = (const float4*)(zj   + (size_t)m * N_DIM);
        const float4* dr4 = (const float4*)(dvec + (size_t)m * N_DIM);
        float4* pw = pkd + (size_t)m * 16;

        float4 zv[8], dv[8];
        float dn2 = 0.f, zjn2 = 0.f;
        #pragma unroll
        for (int q = 0; q < 8; ++q) {
            zv[q] = zr4[q]; dv[q] = dr4[q];
            dn2  = fmaf(dv[q].x,dv[q].x, fmaf(dv[q].y,dv[q].y, fmaf(dv[q].z,dv[q].z, fmaf(dv[q].w,dv[q].w, dn2))));
            zjn2 = fmaf(zv[q].x,zv[q].x, fmaf(zv[q].y,zv[q].y, fmaf(zv[q].z,zv[q].z, fmaf(zv[q].w,zv[q].w, zjn2))));
        }
        float dn  = sqrtf(dn2);
        float inv = (dn > 1e-6f) ? (1.0f / dn) : 0.0f;   // use_proj folded in

        float cj = 0.f;
        #pragma unroll
        for (int q = 0; q < 8; ++q) {
            float4 b4;
            b4.x = dv[q].x * inv; b4.y = dv[q].y * inv;
            b4.z = dv[q].z * inv; b4.w = dv[q].w * inv;
            pw[2*q]   = make_float4(zv[q].x, zv[q].y, b4.x, b4.y);
            pw[2*q+1] = make_float4(zv[q].z, zv[q].w, b4.z, b4.w);
            cj = fmaf(zv[q].x,b4.x, fmaf(zv[q].y,b4.y, fmaf(zv[q].z,b4.z, fmaf(zv[q].w,b4.w, cj))));
        }
        float sp_    = fmaxf(spar[m],  1.1920929e-07f);
        float so_    = fmaxf(sperp[m], 1.1920929e-07f);
        float w_par  = 1.0f / (sp_ * sp_);
        float w_perp = 1.0f / (so_ * so_);
        meta[m] = make_float4(w_perp, w_par - w_perp, cj, zjn2);
    }
    if (m < B_DIM) {
        const float4* zr4 = (const float4*)(z + (size_t)m * N_DIM);
        float acc = 0.f;
        #pragma unroll
        for (int q = 0; q < 8; ++q) {
            float4 a4 = zr4[q];
            acc = fmaf(a4.x,a4.x, fmaf(a4.y,a4.y, fmaf(a4.z,a4.z, fmaf(a4.w,a4.w, acc))));
        }
        zn2[m] = acc;
    }
}

// ---------------------------------------------------------------------------
// Kernel 0b: convert T_hat f32->f16 ONCE and pre-swizzle into the WMMA f16
// B-operand layout, so the main loop fetches each operand as 2x b128.
//   flat u32 index: (((chunk*128 + s)*2 + hi))*8 + v
//     word v = pack(f16 That[32c+16hi+2v][s], f16 That[32c+16hi+2v+1][s])
// ---------------------------------------------------------------------------
__global__ __launch_bounds__(256) void cpsf_pack_that_kernel(
    const float* __restrict__ That, unsigned* __restrict__ ThP)
{
    const int t  = blockIdx.x * 256 + threadIdx.x;   // 0 .. CHUNKS*128*2-1
    const int c  = t >> 8;
    const int s  = (t >> 1) & 127;
    const int hi = t & 1;
    const float* base = That + (size_t)(c * MCH + 16 * hi) * S_DIM + s;

    union { _Float16 h[2]; unsigned u; } pk;
    unsigned w[8];
    #pragma unroll
    for (int v = 0; v < 8; ++v) {
        pk.h[0] = (_Float16)base[(2 * v)     * S_DIM];
        pk.h[1] = (_Float16)base[(2 * v + 1) * S_DIM];
        w[v] = pk.u;
    }
    uint4* dst = (uint4*)(ThP + (size_t)t * 8);
    dst[0] = make_uint4(w[0], w[1], w[2], w[3]);
    dst[1] = make_uint4(w[4], w[5], w[6], w[7]);
}

// ---------------------------------------------------------------------------
// Kernel 1: fused main. Per 32-m chunk:
//   32x v_wmma_f32_16x16x4_f32 (z.zj / z.b), elementwise exp tail,
//   2x v_wmma_f32_16x16x32_f16 (gain @ T_hat) with preconverted B operands.
// ---------------------------------------------------------------------------
__global__ __launch_bounds__(32) void cpsf_main_kernel(
    const float*  __restrict__ z,
    const float*  __restrict__ alpha,
    const float4* __restrict__ pkd,
    const uint4*  __restrict__ ThP4,
    const float4* __restrict__ meta,
    const float*  __restrict__ zn2,
    float* __restrict__ out)
{
    __shared__ _Float16 gt[16 * MCH];    // gain tile [b-row][m-local]

    const int lane = threadIdx.x;        // wave32
    const int b0   = blockIdx.x * 16;
    const int scol = blockIdx.y * 32;

    const int r      = lane & 15;
    const int hi     = lane >> 4;
    const int ncol   = lane & 15;
    const int kbaseA = hi * 16;          // f16 A-operand K base

    // loop-invariant z tile in f32 WMMA A layout: lane holds K = 4q+2hi (+1)
    v2f zA[8];
    {
        const float* zrow = z + (size_t)(b0 + r) * N_DIM + 2 * hi;
        #pragma unroll
        for (int q = 0; q < 8; ++q)
            zA[q] = *(const v2f*)(zrow + 4 * q);
    }
    float zn2w[8];
    #pragma unroll
    for (int g = 0; g < 8; ++g)
        zn2w[g] = zn2[b0 + g + 8 * hi];

    v8f c0 = {}, c1 = {};

    #pragma unroll 1
    for (int m0 = 0; m0 < M_DIM; m0 += MCH) {
        // ---- phase 1: dot-product GEMMs, one b128 feeds both WMMA B ops ----
        v8f P0 = {}, P1 = {}, R0 = {}, R1 = {};
        const float4* pk0 = pkd + (size_t)(m0 + ncol)      * 16 + hi;
        const float4* pk1 = pkd + (size_t)(m0 + 16 + ncol) * 16 + hi;
        #pragma unroll
        for (int q = 0; q < 8; ++q) {
            float4 pv0 = pk0[2 * q];
            float4 pv1 = pk1[2 * q];
            v2f vz0 = {pv0.x, pv0.y}, vb0 = {pv0.z, pv0.w};
            v2f vz1 = {pv1.x, pv1.y}, vb1 = {pv1.z, pv1.w};
            P0 = __builtin_amdgcn_wmma_f32_16x16x4_f32(false, zA[q], false, vz0, (short)0, P0, false, false);
            R0 = __builtin_amdgcn_wmma_f32_16x16x4_f32(false, zA[q], false, vb0, (short)0, R0, false, false);
            P1 = __builtin_amdgcn_wmma_f32_16x16x4_f32(false, zA[q], false, vz1, (short)0, P1, false, false);
            R1 = __builtin_amdgcn_wmma_f32_16x16x4_f32(false, zA[q], false, vb1, (short)0, R1, false, false);
        }

        // ---- elementwise tail on C-layout registers ----
        float4 mt0 = meta[m0 + ncol];
        float4 mt1 = meta[m0 + 16 + ncol];
        float  am0 = alpha[m0 + ncol];
        float  am1 = alpha[m0 + 16 + ncol];
        #pragma unroll
        for (int g = 0; g < 8; ++g) {
            float d0 = zn2w[g] - 2.0f * P0[g] + mt0.w;
            float p0 = R0[g] - mt0.z;
            float q0 = fminf(fmaf(mt0.y, p0 * p0, mt0.x * d0), 25.0f);
            gt[(g + 8 * hi) * MCH + ncol] = (_Float16)(am0 * __expf(-PI_F * q0));

            float d1 = zn2w[g] - 2.0f * P1[g] + mt1.w;
            float p1 = R1[g] - mt1.z;
            float q1 = fminf(fmaf(mt1.y, p1 * p1, mt1.x * d1), 25.0f);
            gt[(g + 8 * hi) * MCH + 16 + ncol] = (_Float16)(am1 * __expf(-PI_F * q1));
        }
        __syncthreads();

        // ---- phase 2 A operand: gains re-read in f16 WMMA A layout ----
        AOp a;
        const unsigned* gw = (const unsigned*)gt;
        #pragma unroll
        for (int v = 0; v < 4; ++v) {
            a.u[v]     = gw[(r * MCH + kbaseA      + 2 * v) >> 1];
            a.u[v + 4] = gw[(r * MCH + kbaseA + 8 + 2 * v) >> 1];
        }

        // ---- phase 2 B operands: preconverted/preswizzled, 2x b128 each ----
        const int  c     = m0 >> 5;                       // chunk index
        const size_t i0  = (size_t)(((c * 128 + scol + ncol)      * 2 + hi)) * 2;
        const size_t i1  = (size_t)(((c * 128 + scol + 16 + ncol) * 2 + hi)) * 2;
        if (m0 + MCH < M_DIM)
            __builtin_prefetch(ThP4 + i0 + 512, 0, 0);    // next chunk = +128*2*2

        AOp bt0, bt1;
        bt0.q4[0] = ThP4[i0];     bt0.q4[1] = ThP4[i0 + 1];
        bt1.q4[0] = ThP4[i1];     bt1.q4[1] = ThP4[i1 + 1];

        c0 = __builtin_amdgcn_wmma_f32_16x16x32_f16(false, a.v, false, bt0.v, (short)0, c0, false, false);
        c1 = __builtin_amdgcn_wmma_f32_16x16x32_f16(false, a.v, false, bt1.v, (short)0, c1, false, false);
        __syncthreads();          // gt reused next iteration
    }

    // ---- store T tiles (16x16 f32 C/D layout) ----
    #pragma unroll
    for (int g = 0; g < 8; ++g) {
        int row = b0 + g + 8 * hi;
        out[(size_t)row * S_DIM + scol + ncol]      = c0[g];
        out[(size_t)row * S_DIM + scol + 16 + ncol] = c1[g];
    }
}

extern "C" void kernel_launch(void* const* d_in, const int* in_sizes, int n_in,
                              void* d_out, int out_size, void* d_ws, size_t ws_size,
                              hipStream_t stream) {
    (void)in_sizes; (void)n_in; (void)out_size; (void)ws_size;
    const float* z     = (const float*)d_in[0];
    const float* zj    = (const float*)d_in[1];
    const float* dvec  = (const float*)d_in[2];
    const float* That  = (const float*)d_in[3];
    const float* alpha = (const float*)d_in[4];
    const float* spar  = (const float*)d_in[5];
    const float* sperp = (const float*)d_in[6];
    float* out = (float*)d_out;

    char* ws = (char*)d_ws;
    float4*   pkd  = (float4*)(ws + WS_PKD);
    unsigned* ThP  = (unsigned*)(ws + WS_THP);
    float4*   meta = (float4*)(ws + WS_META);
    float*    zn2  = (float*)(ws + WS_ZN2);

    cpsf_precompute_kernel<<<dim3(M_DIM / 256), dim3(256), 0, stream>>>(
        z, zj, dvec, spar, sperp, pkd, meta, zn2);

    cpsf_pack_that_kernel<<<dim3(CHUNKS * 128 * 2 / 256), dim3(256), 0, stream>>>(
        That, ThP);

    cpsf_main_kernel<<<dim3(B_DIM / 16, S_DIM / 32), dim3(32), 0, stream>>>(
        z, alpha, pkd, (const uint4*)ThP, meta, zn2, out);
}
